// LSTM_55860344652279
// MI455X (gfx1250) — matrix operand order repaired and checked by reading, not compiled, not run on hardware
//
#include <hip/hip_runtime.h>
#include <hip/hip_bf16.h>
#include <math.h>

typedef __attribute__((ext_vector_type(16))) __bf16 v16bf;
typedef __attribute__((ext_vector_type(8)))  __bf16 v8bf;
typedef __attribute__((ext_vector_type(8)))  float  v8f;

#define BSZ  512
#define TLEN 512
#define INF  128
#define HID  256

// ---------------------------------------------------------------------------
// A-operand loader (16-bit WMMA A layout, 16x32 tile) from row-major LDS.
// lanes 0-15 : row = lane,     K[k0+0..15]
// lanes 16-31: row = lane-16,  K[k0+16..31]
// ---------------------------------------------------------------------------
__device__ __forceinline__ v16bf ldsA(const __bf16* base, int ldk, int k0, int lane) {
    const int row  = lane & 15;
    const int koff = k0 + (lane & 16);              // +16 halves for upper half-wave
    const __bf16* p = base + row * ldk + koff;
    union { v16bf v; v8bf h[2]; } u;
    u.h[0] = *reinterpret_cast<const v8bf*>(p);      // 16B LDS load
    u.h[1] = *reinterpret_cast<const v8bf*>(p + 8);  // 16B LDS load
    return u.v;
}

// ---------------------------------------------------------------------------
// B-operand loader (16-bit WMMA B layout, 32x16 tile) from [N][K] row-major
// global (pre-transposed weights). lane holds column n=lane%16, contiguous
// 16 K-values starting at k0 (+16 for upper half-wave) -> one 32B load.
// 32-bit offset arithmetic so the compiler emits global_load saddr form
// (uniform 64-bit base in SGPRs + per-lane 32-bit offset, LOADcnt only).
// ---------------------------------------------------------------------------
__device__ __forceinline__ v16bf glbB(const __bf16* gbase, int ldk, int n_base,
                                      int k0, int lane) {
    const int n   = lane & 15;
    const int kk  = k0 + (lane & 16);
    const int off = (n_base + n) * ldk + kk;        // elements, < 2^20
    return *reinterpret_cast<const v16bf*>(gbase + off);
}

__device__ __forceinline__ float sigmoidf_(float x) {
    return 1.0f / (1.0f + __expf(-x));
}
// tanh(x) = 2*sigmoid(2x) - 1 : one v_exp_f32 TRANS op instead of libm tanhf
__device__ __forceinline__ float tanhf_fast(float x) {
    return 2.0f / (1.0f + __expf(-2.0f * x)) - 1.0f;
}

// ---------------------------------------------------------------------------
// Prep: convert fp32 weights -> bf16, transposed to [N][K] for WMMA B operand.
//   WxT: 4 gates x [HID][IN]   (gate order i,f,o,c)
//   WhT: 4 gates x [HID][HID]
// ---------------------------------------------------------------------------
__global__ void lstm_prep_weights(const float* __restrict__ Wxi, const float* __restrict__ Wxf,
                                  const float* __restrict__ Wxo, const float* __restrict__ Wxc,
                                  const float* __restrict__ Whi, const float* __restrict__ Whf,
                                  const float* __restrict__ Who, const float* __restrict__ Whc,
                                  __bf16* __restrict__ WxT, __bf16* __restrict__ WhT) {
    int idx = blockIdx.x * blockDim.x + threadIdx.x;
    const int nWx = 4 * HID * INF;
    if (idx < nWx) {
        int g = idx / (HID * INF);
        int rem = idx - g * (HID * INF);
        int n = rem / INF, k = rem - n * INF;
        const float* src = (g == 0) ? Wxi : (g == 1) ? Wxf : (g == 2) ? Wxo : Wxc;
        WxT[idx] = (__bf16)src[k * HID + n];
        return;
    }
    idx -= nWx;
    if (idx < 4 * HID * HID) {
        int g = idx / (HID * HID);
        int rem = idx - g * (HID * HID);
        int n = rem / HID, k = rem - n * HID;
        const float* src = (g == 0) ? Whi : (g == 1) ? Whf : (g == 2) ? Who : Whc;
        WhT[idx] = (__bf16)src[k * HID + n];
    }
}

// ---------------------------------------------------------------------------
// Persistent LSTM kernel: each WG owns 32 batch rows for all T steps.
// 16 waves; wave w owns hidden columns [16w, 16w+16) and both 16-row m-tiles,
// computing all 4 gates itself so the nonlinearity stays in registers.
// H state lives in LDS (bf16), C state in registers (fp32).
// Weight B-operands are streamed from L2 every step; the laundered integer
// offset below keeps those loads inside the t-loop (prevents LICM -> spill)
// while preserving the global address space (global_load, LOADcnt only).
// ---------------------------------------------------------------------------
__global__ __launch_bounds__(512)
void lstm_main_kernel(const float* __restrict__ seq,
                      const __bf16* __restrict__ WxT, const __bf16* __restrict__ WhT,
                      const float* __restrict__ b_i, const float* __restrict__ b_f,
                      const float* __restrict__ b_o, const float* __restrict__ b_c,
                      float* __restrict__ Hout, float* __restrict__ Cout) {
    __shared__ __bf16 Xs[32][INF];   //  8 KB: current-step inputs (bf16)
    __shared__ __bf16 Hs[32][HID];   // 16 KB: hidden state (bf16)

    const int tid    = threadIdx.x;
    const int lane   = tid & 31;
    const int wave   = tid >> 5;           // 0..15 => n-tile
    const int n_base = wave * 16;
    const int b_base = blockIdx.x * 32;
    const int ncol   = n_base + (lane & 15);
    const int m_hi   = (lane & 16) >> 1;   // +8 row offset for upper half-wave

    // zero initial hidden state
    for (int i = tid; i < 32 * HID; i += 512) (&Hs[0][0])[i] = (__bf16)0.0f;

    float biasv[4];
    biasv[0] = b_i[ncol]; biasv[1] = b_f[ncol];
    biasv[2] = b_o[ncol]; biasv[3] = b_c[ncol];

    float Creg[2][8], Hreg[2][8];
#pragma unroll
    for (int m = 0; m < 2; ++m)
#pragma unroll
        for (int r = 0; r < 8; ++r) Creg[m][r] = 0.0f;

    int wofs = 0;   // laundered each iteration; always 0 at runtime

#pragma unroll 1
    for (int t = 0; t < TLEN; ++t) {
        // Opaque, loop-carried offset: the 48 B-operand loads per step depend
        // on it, so they cannot be hoisted out of the t-loop (which would need
        // 384 VGPRs of weights and spill to scratch). Weights stay resident in
        // the 192MB L2; re-reading them each step is nearly free. Laundering an
        // int (not the pointer) preserves addrspace(1) -> global_load saddr.
        asm volatile("" : "+s"(wofs));
        const __bf16* wxp = WxT + wofs;
        const __bf16* whp = WhT + wofs;

        __syncthreads();   // Hs writes of prev step visible; prev Xs reads done
        // cooperative load of X_t tile: 32 rows x 128 cols fp32 -> bf16
        for (int i = tid; i < (32 * INF) / 4; i += 512) {
            const int row = (i * 4) / INF, col = (i * 4) % INF;
            const float* src = &seq[((size_t)(b_base + row) * TLEN + t) * INF + col];
            const float4 x4 = *reinterpret_cast<const float4*>(src);
            Xs[row][col + 0] = (__bf16)x4.x;
            Xs[row][col + 1] = (__bf16)x4.y;
            Xs[row][col + 2] = (__bf16)x4.z;
            Xs[row][col + 3] = (__bf16)x4.w;
            if (t + 1 < TLEN) __builtin_prefetch(src + INF, 0, 1); // next step's X
        }
        __syncthreads();

        v8f acc[4][2];
#pragma unroll
        for (int g = 0; g < 4; ++g)
#pragma unroll
            for (int m = 0; m < 2; ++m)
#pragma unroll
                for (int r = 0; r < 8; ++r) acc[g][m][r] = biasv[g];

        // input projection: K = 128, 4 chunks of 32
#pragma unroll
        for (int k0 = 0; k0 < INF; k0 += 32) {
            const v16bf a0 = ldsA(&Xs[0][0],  INF, k0, lane);
            const v16bf a1 = ldsA(&Xs[16][0], INF, k0, lane);
#pragma unroll
            for (int g = 0; g < 4; ++g) {
                const v16bf bm = glbB(wxp + g * HID * INF, INF, n_base, k0, lane);
                acc[g][0] = __builtin_amdgcn_wmma_f32_16x16x32_bf16(
                    false, a0, false, bm, (short)0, acc[g][0], false, false);
                acc[g][1] = __builtin_amdgcn_wmma_f32_16x16x32_bf16(
                    false, a1, false, bm, (short)0, acc[g][1], false, false);
            }
        }
        // hidden projection: K = 256, 8 chunks of 32
#pragma unroll
        for (int k0 = 0; k0 < HID; k0 += 32) {
            const v16bf a0 = ldsA(&Hs[0][0],  HID, k0, lane);
            const v16bf a1 = ldsA(&Hs[16][0], HID, k0, lane);
#pragma unroll
            for (int g = 0; g < 4; ++g) {
                const v16bf bm = glbB(whp + g * HID * HID, HID, n_base, k0, lane);
                acc[g][0] = __builtin_amdgcn_wmma_f32_16x16x32_bf16(
                    false, a0, false, bm, (short)0, acc[g][0], false, false);
                acc[g][1] = __builtin_amdgcn_wmma_f32_16x16x32_bf16(
                    false, a1, false, bm, (short)0, acc[g][1], false, false);
            }
        }

        // gate nonlinearities + state update (all in-register)
#pragma unroll
        for (int m = 0; m < 2; ++m)
#pragma unroll
            for (int r = 0; r < 8; ++r) {
                const float I  = sigmoidf_(acc[0][m][r]);
                const float F  = sigmoidf_(acc[1][m][r]);
                const float O  = sigmoidf_(acc[2][m][r]);
                const float Ct = tanhf_fast(acc[3][m][r]);
                const float Cn = F * Creg[m][r] + I * Ct;
                Creg[m][r] = Cn;
                Hreg[m][r] = O * tanhf_fast(Cn);
            }

        __syncthreads();   // everyone finished reading old Hs
        // scatter new H (bf16) back to LDS per C/D layout: M = r + 8*(lane>=16)
#pragma unroll
        for (int m = 0; m < 2; ++m)
#pragma unroll
            for (int r = 0; r < 8; ++r)
                Hs[m * 16 + m_hi + r][ncol] = (__bf16)Hreg[m][r];
    }

    // final H and C to global (fp32)
#pragma unroll
    for (int m = 0; m < 2; ++m)
#pragma unroll
        for (int r = 0; r < 8; ++r) {
            const int grow = b_base + m * 16 + m_hi + r;
            Hout[(size_t)grow * HID + ncol] = Hreg[m][r];
            Cout[(size_t)grow * HID + ncol] = Creg[m][r];
        }
}

// ---------------------------------------------------------------------------
// Epilogue: Y[b] = H[b,:] . W_hq[:,0] + b_q   (one block per batch row)
// ---------------------------------------------------------------------------
__global__ __launch_bounds__(256)
void lstm_y_kernel(const float* __restrict__ Hout, const float* __restrict__ W_hq,
                   const float* __restrict__ b_q, float* __restrict__ Y) {
    __shared__ float red[256];
    const int b = blockIdx.x;
    red[threadIdx.x] = Hout[(size_t)b * HID + threadIdx.x] * W_hq[threadIdx.x];
    __syncthreads();
#pragma unroll
    for (int s = 128; s > 0; s >>= 1) {
        if (threadIdx.x < s) red[threadIdx.x] += red[threadIdx.x + s];
        __syncthreads();
    }
    if (threadIdx.x == 0) Y[b] = red[0] + b_q[0];
}

extern "C" void kernel_launch(void* const* d_in, const int* in_sizes, int n_in,
                              void* d_out, int out_size, void* d_ws, size_t ws_size,
                              hipStream_t stream) {
    const float* seq  = (const float*)d_in[0];
    const float* W_xi = (const float*)d_in[1];
    const float* W_hi = (const float*)d_in[2];
    const float* b_i  = (const float*)d_in[3];
    const float* W_xf = (const float*)d_in[4];
    const float* W_hf = (const float*)d_in[5];
    const float* b_f  = (const float*)d_in[6];
    const float* W_xo = (const float*)d_in[7];
    const float* W_ho = (const float*)d_in[8];
    const float* b_o  = (const float*)d_in[9];
    const float* W_xc = (const float*)d_in[10];
    const float* W_hc = (const float*)d_in[11];
    const float* b_c  = (const float*)d_in[12];
    const float* W_hq = (const float*)d_in[13];
    const float* b_q  = (const float*)d_in[14];

    __bf16* WxT = (__bf16*)d_ws;                                  // 4*256*128 bf16
    __bf16* WhT = (__bf16*)((char*)d_ws + (size_t)4 * HID * INF * sizeof(__bf16));

    float* Y    = (float*)d_out;           // [512]
    float* Hout = Y + BSZ;                 // [512,256]
    float* Cout = Hout + (size_t)BSZ * HID;// [512,256]

    const int nPrep = 4 * HID * INF + 4 * HID * HID;
    lstm_prep_weights<<<(nPrep + 255) / 256, 256, 0, stream>>>(
        W_xi, W_xf, W_xo, W_xc, W_hi, W_hf, W_ho, W_hc, WxT, WhT);

    lstm_main_kernel<<<BSZ / 32, 512, 0, stream>>>(
        seq, WxT, WhT, b_i, b_f, b_o, b_c, Hout, Cout);

    lstm_y_kernel<<<BSZ, 256, 0, stream>>>(Hout, W_hq, b_q, Y);
}
